// EMATranVectorQuantizer_35459249996162
// MI455X (gfx1250) — matrix operand order repaired and checked by reading, not compiled, not run on hardware
//
#include <hip/hip_runtime.h>
#include <hip/hip_bf16.h>

#define K_CODES 128
#define DIM 32
#define WSTRIDE 34     // f32 stats rows: 34 mod 64 banks -> conflict-free, 8B-even
#define CBH_STRIDE 40  // f16 codebook rows: 20 dwords -> conflict-free, 80B keeps 16B align
#define WAVES 8
#define TPW 8          // 16-row tiles per wave

typedef float    __attribute__((ext_vector_type(4)))  v4f;
typedef float    __attribute__((ext_vector_type(8)))  v8f;
typedef _Float16 __attribute__((ext_vector_type(8)))  v8h;
typedef _Float16 __attribute__((ext_vector_type(16))) v16h;

__device__ __forceinline__ v16h cat16(v8h lo, v8h hi) {
  v16h r;
#pragma unroll
  for (int j = 0; j < 8; ++j) { r[j] = lo[j]; r[8 + j] = hi[j]; }
  return r;
}

__device__ __forceinline__ v8f wmma_f16(v16h a, v16h b, v8f c) {
  // D = A(16x32 f16) x B(32x16 f16) + C(16x16 f32)  -> v_wmma_f32_16x16x32_f16
  return __builtin_amdgcn_wmma_f32_16x16x32_f16(false, a, false, b, (short)0, c, false, false);
}

// ---------------------------------------------------------------- init: zero stats accumulators in d_ws
__global__ void vq_init(float* __restrict__ acc) {
  int i = blockIdx.x * blockDim.x + threadIdx.x;
  if (i < K_CODES * (DIM + 1)) acc[i] = 0.0f;   // [0,128)=cnt, [128,128+4096)=w
}

// ---------------------------------------------------------------- main: distances via split-f16 WMMA + argmin + stats
__global__ __launch_bounds__(256) void vq_main(const float* __restrict__ latent,
                                               const float* __restrict__ codebook,
                                               float* __restrict__ outPolicy,
                                               float* __restrict__ outQuant,
                                               float* __restrict__ gCnt,
                                               float* __restrict__ gW,
                                               long long totalTiles) {
  __shared__ alignas(64) _Float16 sCBh[K_CODES * CBH_STRIDE];  // codebook hi (f16)
  __shared__ alignas(64) _Float16 sCBl[K_CODES * CBH_STRIDE];  // codebook lo (f16 residual)
  __shared__ float sCBf[K_CODES * DIM];                        // exact f32 codebook for gather
  __shared__ float sCnorm[K_CODES];
  __shared__ float sW[K_CODES * WSTRIDE];
  __shared__ float sCnt[K_CODES];

  const int tid  = threadIdx.x;
  const int lane = tid & 31;
  const int wave = tid >> 5;

  // Stage codebook: f32 copy + hi/lo f16 split. Zero block-local stats.
  for (int j = tid; j < K_CODES * DIM; j += 256) {
    const int c = j >> 5, d = j & 31;
    const float f = codebook[j];
    const _Float16 h = (_Float16)f;
    sCBf[j] = f;
    sCBh[c * CBH_STRIDE + d] = h;
    sCBl[c * CBH_STRIDE + d] = (_Float16)(f - (float)h);
  }
  for (int j = tid; j < K_CODES * WSTRIDE; j += 256) sW[j] = 0.0f;
  if (tid < K_CODES) sCnt[tid] = 0.0f;
  __syncthreads();
  if (tid < K_CODES) {
    float s = 0.0f;
#pragma unroll
    for (int d = 0; d < DIM; ++d) { float c = sCBf[tid * DIM + d]; s += c * c; }
    sCnorm[tid] = s;
  }
  __syncthreads();

  const int m     = lane & 15;   // row (A) / column (B,C,D) index within 16
  const int khalf = lane >> 4;   // half-select per ISA 16-bit layouts

  for (int t = 0; t < TPW; ++t) {
    const long long tile = (long long)blockIdx.x * (WAVES * TPW) + wave * TPW + t;
    if (tile >= totalTiles) break;
    const long long rowBase = tile * 16;

    // A fragment: lane holds A[m][8k+0..7] and A[m][16+8k+0..7]  (k = khalf)
    const float* arow = latent + (rowBase + m) * DIM + khalf * 8;
    v4f x0 = *(const v4f*)(arow +  0);
    v4f x1 = *(const v4f*)(arow +  4);
    v4f x2 = *(const v4f*)(arow + 16);
    v4f x3 = *(const v4f*)(arow + 20);
    float xa[16];
#pragma unroll
    for (int j = 0; j < 4; ++j) { xa[j] = x0[j]; xa[4 + j] = x1[j]; xa[8 + j] = x2[j]; xa[12 + j] = x3[j]; }
    v16h ahi, alo;
#pragma unroll
    for (int j = 0; j < 16; ++j) {
      const _Float16 h = (_Float16)xa[j];
      ahi[j] = h;
      alo[j] = (_Float16)(xa[j] - (float)h);
    }

    float bestVal[8];
    int   bestIdx[8];
#pragma unroll
    for (int r = 0; r < 8; ++r) { bestVal[r] = 3.4e38f; bestIdx[r] = 0; }

#pragma unroll
    for (int ct = 0; ct < 8; ++ct) {           // 8 code tiles of 16 codes
      const int code = ct * 16 + m;            // this lane's column
      // B fragment: lane holds cb[code][8k+0..7] and cb[code][16+8k+0..7]  (column-major B = codebook.T)
      const _Float16* bh = sCBh + code * CBH_STRIDE + khalf * 8;
      const _Float16* bl = sCBl + code * CBH_STRIDE + khalf * 8;
      const v16h bhi = cat16(*(const v8h*)(bh), *(const v8h*)(bh + 16));
      const v16h blo = cat16(*(const v8h*)(bl), *(const v8h*)(bl + 16));

      v8f acc = {};
      acc = wmma_f16(ahi, bhi, acc);   // hi*hi
      acc = wmma_f16(alo, bhi, acc);   // lo*hi
      acc = wmma_f16(ahi, blo, acc);   // hi*lo  -> ~22-bit accurate dot product
      const float cn = sCnorm[code];
#pragma unroll
      for (int r = 0; r < 8; ++r) {            // rows r / r+8 per lane-half
        float sc = cn - 2.0f * acc[r];         // ||x||^2 constant per row -> dropped
        if (sc < bestVal[r] || (sc == bestVal[r] && code < bestIdx[r])) {
          bestVal[r] = sc; bestIdx[r] = code;
        }
      }
    }

    // Min+index butterfly within each 16-lane half (masks stay inside the half).
#pragma unroll
    for (int r = 0; r < 8; ++r) {
#pragma unroll
      for (int msk = 1; msk < 16; msk <<= 1) {
        float ov = __shfl_xor(bestVal[r], msk, 32);
        int   oi = __shfl_xor(bestIdx[r], msk, 32);
        if (ov < bestVal[r] || (ov == bestVal[r] && oi < bestIdx[r])) {
          bestVal[r] = ov; bestIdx[r] = oi;
        }
      }
    }

    // Emit quantized / policy rows (coalesced, 32 lanes = 32 dims) + LDS stats.
#pragma unroll
    for (int i = 0; i < 16; ++i) {
      const int idx = __shfl(bestIdx[i & 7], (i < 8) ? 0 : 16, 32);
      const long long row = rowBase + i;
      const float l = latent[row * DIM + lane];
      const float q = sCBf[idx * DIM + lane];        // exact f32 codebook row
      outQuant[row * DIM + lane]  = q;
      outPolicy[row * DIM + lane] = l + (q - l);     // straight-through forward value
      atomicAdd(&sW[idx * WSTRIDE + lane], l);       // ds_add_f32
      if (lane == 0) atomicAdd(&sCnt[idx], 1.0f);
    }
  }

  __syncthreads();
  // One global flush per block.
  for (int j = tid; j < K_CODES * DIM; j += 256)
    atomicAdd(&gW[j], sW[(j >> 5) * WSTRIDE + (j & 31)]);
  if (tid < K_CODES) atomicAdd(&gCnt[tid], sCnt[tid]);
}

// ---------------------------------------------------------------- finalize: EMA update + Laplace-smoothed codebook
__global__ void vq_final(const float* __restrict__ gCnt, const float* __restrict__ gW,
                         const float* __restrict__ emaCS, const float* __restrict__ emaW,
                         float* __restrict__ outCB, float* __restrict__ outCS,
                         float* __restrict__ outW) {
  __shared__ float red[K_CODES];
  const int t = threadIdx.x;              // 128 threads
  const float decay = 0.99f;
  const float omd   = (float)(1.0 - 0.99);
  float ncs = decay * emaCS[t] + omd * gCnt[t];
  outCS[t] = ncs;
  red[t] = ncs;
  __syncthreads();
  for (int s = 64; s > 0; s >>= 1) {
    if (t < s) red[t] += red[t + s];
    __syncthreads();
  }
  const float n = red[0];
  const float cluster = (ncs + 1e-5f) / (n + (float)K_CODES * 1e-5f) * n;
#pragma unroll
  for (int d = 0; d < DIM; ++d) {
    float w = decay * emaW[t * DIM + d] + omd * gW[t * DIM + d];
    outW[t * DIM + d]  = w;
    outCB[t * DIM + d] = w / cluster;
  }
}

// ----------------------------------------------------------------
extern "C" void kernel_launch(void* const* d_in, const int* in_sizes, int n_in,
                              void* d_out, int out_size, void* d_ws, size_t ws_size,
                              hipStream_t stream) {
  (void)n_in; (void)out_size; (void)ws_size;
  const float* latent   = (const float*)d_in[0];
  const float* codebook = (const float*)d_in[1];
  const float* emaCS    = (const float*)d_in[2];
  const float* emaW     = (const float*)d_in[3];

  float* O = (float*)d_out;
  const long long NL = (long long)in_sizes[0];          // 131072*8*32 = 33,554,432
  float* outPolicy = O;
  float* outQuant  = O + NL;
  float* outCB     = O + 2 * NL;
  float* outCS     = O + 2 * NL + (long long)K_CODES * DIM;
  float* outW      = O + 2 * NL + (long long)K_CODES * DIM + K_CODES;

  float* acc  = (float*)d_ws;       // 4224 floats
  float* gCnt = acc;
  float* gW   = acc + K_CODES;

  vq_init<<<(K_CODES * (DIM + 1) + 255) / 256, 256, 0, stream>>>(acc);

  const long long nvec       = NL / DIM;        // 1,048,576
  const long long totalTiles = (nvec + 15) / 16;
  const long long tilesPerBlock = WAVES * TPW;  // 64
  const int blocks = (int)((totalTiles + tilesPerBlock - 1) / tilesPerBlock);  // 1024
  vq_main<<<blocks, 256, 0, stream>>>(latent, codebook, outPolicy, outQuant, gCnt, gW, totalTiles);

  vq_final<<<1, K_CODES, 0, stream>>>(gCnt, gW, emaCS, emaW, outCB, outCS, outW);
}
